// MultiHeadedSelfAttention_50792283243220
// MI455X (gfx1250) — compile-verified, matrix operand order checked by reading, not run on hardware
//
#include <hip/hip_runtime.h>
#include <math.h>

typedef unsigned short u16;
typedef __attribute__((ext_vector_type(16))) __bf16 v16bf;
typedef __attribute__((ext_vector_type(8)))  float  v8f;
typedef __attribute__((ext_vector_type(4)))  int    v4i;

#define B_DIM 4
#define S_LEN 2048
#define D_DIM 1024
#define H_CNT 16
#define W_HD  64
#define M_TOT (B_DIM * S_LEN)   // 8192

#define GLOBAL_AS __attribute__((address_space(1)))
#define LDS_AS    __attribute__((address_space(3)))

#if defined(__has_builtin)
#if __has_builtin(__builtin_amdgcn_global_load_async_to_lds_b128)
#define HAS_ASYNC_LDS 1
#else
#define HAS_ASYNC_LDS 0
#endif
#else
#define HAS_ASYNC_LDS 0
#endif

// ---------- helpers ----------

__device__ __forceinline__ u16 f2bf(float f) {
  unsigned u = __float_as_uint(f);
  u += 0x7FFFu + ((u >> 16) & 1u);     // round to nearest even
  return (u16)(u >> 16);
}

__device__ __forceinline__ v8f zero8() {
  v8f z = {0.f, 0.f, 0.f, 0.f, 0.f, 0.f, 0.f, 0.f};
  return z;
}

// 16-byte global -> LDS copy. Uses the gfx1250 async DMA path when the
// builtin exists (GLOBAL_LOAD_ASYNC_TO_LDS_B128, tracked by ASYNCcnt);
// otherwise a plain register round-trip.
// Probe-derived signature: (int4 AS1*, int4 AS3*, imm offset, imm cpol).
__device__ __forceinline__ void copy16_g2l(const u16* g, u16* l) {
#if HAS_ASYNC_LDS
  __builtin_amdgcn_global_load_async_to_lds_b128(
      (GLOBAL_AS v4i*)(g), (LDS_AS v4i*)(l), 0, 0);
#else
  *reinterpret_cast<uint4*>(l) = *reinterpret_cast<const uint4*>(g);
#endif
}

__device__ __forceinline__ void wait_async_lds() {
#if HAS_ASYNC_LDS
#if __has_builtin(__builtin_amdgcn_s_wait_asynccnt)
  __builtin_amdgcn_s_wait_asynccnt(0);
#else
  asm volatile("s_wait_asynccnt 0" ::: "memory");
#endif
#endif
}

// Load a 16x32 bf16 fragment (A layout per CDNA5 ISA 7.12.2) from a
// row-major matrix with leading dimension ldr (in elements).
//   lane<16 : row = lane,     K in {0..7} U {16..23}
//   lane>=16: row = lane-16,  K in {8..15} U {24..31}
// B fragments use the same pattern from (N,K)-row-major storage
// (lane = output column, K striped across VGPRs).
__device__ __forceinline__ v16bf load_frag(const u16* base, int ldr) {
  int lane = threadIdx.x & 31;
  int row  = lane & 15;
  int kb   = (lane >> 4) << 3;   // 0 or 8
  union { v16bf v; uint4 q[2]; } u;
  u.q[0] = *reinterpret_cast<const uint4*>(base + (size_t)row * ldr + kb);
  u.q[1] = *reinterpret_cast<const uint4*>(base + (size_t)row * ldr + 16 + kb);
  return u.v;
}

__device__ __forceinline__ v8f wmma_bf16(v16bf a, v16bf b, v8f c) {
  return __builtin_amdgcn_wmma_f32_16x16x32_bf16(
      /*neg_a=*/false, a, /*neg_b=*/false, b,
      /*c_mod=*/(short)0, c, /*reuse_a=*/false, /*reuse_b=*/false);
}

// reductions across 16-lane halves (C-layout rows live in 16-lane halves)
__device__ __forceinline__ float hmax16(float x) {
#pragma unroll
  for (int m = 1; m < 16; m <<= 1) x = fmaxf(x, __shfl_xor(x, m, 32));
  return x;
}
__device__ __forceinline__ float hadd16(float x) {
#pragma unroll
  for (int m = 1; m < 16; m <<= 1) x += __shfl_xor(x, m, 32);
  return x;
}

// ---------- kernels ----------

__global__ void cast_f32_to_bf16(const float* __restrict__ src,
                                 u16* __restrict__ dst, int n) {
  for (int i = blockIdx.x * blockDim.x + threadIdx.x; i < n;
       i += gridDim.x * blockDim.x)
    dst[i] = f2bf(src[i]);
}

// Projection GEMM:  Out[b,h,s,d] = sum_k X[(b,s),k] * W[n,k] + bias[n].
// Block = 256 threads (8 waves).  Block tile 128(M) x 64(N), K step 32.
// The weight tile (shared by all 8 waves) is staged in LDS, double-buffered,
// via async-to-LDS DMA; A fragments are register double-buffered.
__global__ __launch_bounds__(256) void proj_wmma(
    const u16* __restrict__ X, const u16* __restrict__ W,
    const float* __restrict__ bias, u16* __restrict__ Out) {
  __shared__ u16 Bt[2][64 * 32];   // [buf][n(64) x k(32)] row-major (N,K)

  const int wave = threadIdx.x >> 5;
  const int lane = threadIdx.x & 31;
  const int m0 = blockIdx.x * 128 + wave * 16;
  const int n0 = blockIdx.y * 64;

  // 64x32 u16 tile = 4 KiB = 256 x 16B chunks; one chunk per thread.
  auto stageB = [&](int buf, int kk) {
    const int c  = threadIdx.x;
    const int n  = c >> 2;
    const int ko = (c & 3) << 3;
    copy16_g2l(W + (size_t)(n0 + n) * D_DIM + kk + ko, &Bt[buf][n * 32 + ko]);
  };

  v8f acc[4] = {zero8(), zero8(), zero8(), zero8()};

  stageB(0, 0);
  v16bf a_cur = load_frag(X + (size_t)m0 * D_DIM, D_DIM);

  int buf = 0;
  for (int kk = 0; kk < D_DIM; kk += 32) {
    wait_async_lds();
    __syncthreads();                     // tile[buf] ready; tile[buf^1] free
    const bool more = (kk + 32 < D_DIM);
    if (more) stageB(buf ^ 1, kk + 32);  // overlap DMA with math
    v16bf a_next = a_cur;
    if (more) a_next = load_frag(X + (size_t)m0 * D_DIM + kk + 32, D_DIM);

    v16bf b0 = load_frag(&Bt[buf][0 * 32],  32);
    v16bf b1 = load_frag(&Bt[buf][16 * 32], 32);
    v16bf b2 = load_frag(&Bt[buf][32 * 32], 32);
    v16bf b3 = load_frag(&Bt[buf][48 * 32], 32);
    acc[0] = wmma_bf16(a_cur, b0, acc[0]);
    acc[1] = wmma_bf16(a_cur, b1, acc[1]);
    acc[2] = wmma_bf16(a_cur, b2, acc[2]);
    acc[3] = wmma_bf16(a_cur, b3, acc[3]);

    a_cur = a_next;
    buf ^= 1;
  }

  // Epilogue: add bias, cast to bf16, scatter into head-major (b,h,s,d).
  const int half = (lane >> 4) << 3;
#pragma unroll
  for (int t = 0; t < 4; ++t) {
    const int n = n0 + t * 16 + (lane & 15);
    const float bv = bias[n];
    const int h = n >> 6, d = n & 63;
#pragma unroll
    for (int r = 0; r < 8; ++r) {
      const int m = m0 + r + half;
      const int bb = m >> 11, s = m & 2047;
      Out[((((size_t)bb * H_CNT + h) * S_LEN) + s) * W_HD + d] =
          f2bf(acc[t][r] + bv);
    }
  }
}

// Flash attention.  Q/K/V: bf16, head-major (bh, s, 64).
// Block = 256 threads (8 waves); block owns (bh, 128 query rows);
// wave owns 16 query rows.  K/V tiles (64 keys) double-buffered in LDS:
// K staged via async-to-LDS DMA, V staged with an on-the-fly 16-bit transpose.
__global__ __launch_bounds__(256) void attn_flash_wmma(
    const u16* __restrict__ Q, const u16* __restrict__ K,
    const u16* __restrict__ V, float* __restrict__ out) {
  __shared__ u16 Kt[2][64 * 64];     // keys, row-major [t][d]
  __shared__ u16 Vt[2][64 * 64];     // values transposed [d][t]
  __shared__ u16 Pt[8][16 * 64];     // per-wave P staging, row-major [m][t]

  const int bh   = blockIdx.x;
  const int wave = threadIdx.x >> 5;
  const int lane = threadIdx.x & 31;
  const int bb = bh >> 4, h = bh & 15;
  const int q0 = blockIdx.y * 128 + wave * 16;

  const u16* Qbh = Q + (size_t)bh * S_LEN * W_HD;
  const u16* Kbh = K + (size_t)bh * S_LEN * W_HD;
  const u16* Vbh = V + (size_t)bh * S_LEN * W_HD;

  // 64x64 u16 tile = 8 KiB = 512 x 16B chunks; 2 chunks per thread.
  auto stageK = [&](int buf, int t0) {
#pragma unroll
    for (int c = threadIdx.x; c < 512; c += 256) {
      const int row = c >> 3, off = (c & 7) << 3;
      copy16_g2l(Kbh + (size_t)(t0 + row) * W_HD + off, &Kt[buf][row * 64 + off]);
    }
  };
  auto stageV = [&](int buf, int t0) {
#pragma unroll
    for (int c = threadIdx.x; c < 512; c += 256) {
      const int row = c >> 3, off = (c & 7) << 3;
      union { uint4 q; u16 s[8]; } uv;
      uv.q = *reinterpret_cast<const uint4*>(&Vbh[(size_t)(t0 + row) * W_HD + off]);
#pragma unroll
      for (int j = 0; j < 8; ++j) Vt[buf][(off + j) * 64 + row] = uv.s[j];
    }
  };

  // Q fragments (d 0..31, 32..63) live in registers for the whole loop.
  const v16bf aq0 = load_frag(Qbh + (size_t)q0 * W_HD + 0, W_HD);
  const v16bf aq1 = load_frag(Qbh + (size_t)q0 * W_HD + 32, W_HD);

  float mrow[8], lrow[8];
  v8f o[4] = {zero8(), zero8(), zero8(), zero8()};
#pragma unroll
  for (int r = 0; r < 8; ++r) { mrow[r] = -INFINITY; lrow[r] = 0.f; }

  const float scale = 0.125f;  // 1/sqrt(64)
  const int half = (lane >> 4) << 3;

  stageK(0, 0);
  stageV(0, 0);

  int buf = 0;
  for (int t0 = 0; t0 < S_LEN; t0 += 64) {
    wait_async_lds();
    __syncthreads();                       // tiles[buf] ready; [buf^1] free
    if (t0 + 64 < S_LEN) {                 // prefetch next tiles during math
      stageK(buf ^ 1, t0 + 64);
      stageV(buf ^ 1, t0 + 64);
    }

    // ---- scores S = Q(16x64) @ K^T(64x64): load all frags, then wmma burst
    v16bf kb[8];
#pragma unroll
    for (int t = 0; t < 4; ++t) {
      kb[2 * t]     = load_frag(&Kt[buf][(t * 16) * 64 + 0], 64);
      kb[2 * t + 1] = load_frag(&Kt[buf][(t * 16) * 64 + 32], 64);
    }
    v8f sc[4] = {zero8(), zero8(), zero8(), zero8()};
#pragma unroll
    for (int t = 0; t < 4; ++t) {
      sc[t] = wmma_bf16(aq0, kb[2 * t], sc[t]);
      sc[t] = wmma_bf16(aq1, kb[2 * t + 1], sc[t]);
    }

    // ---- online softmax (rows live across 16-lane halves)
    float alpha[8];
#pragma unroll
    for (int r = 0; r < 8; ++r) {
      float mx = fmaxf(fmaxf(sc[0][r], sc[1][r]), fmaxf(sc[2][r], sc[3][r]));
      mx = hmax16(mx * scale);
      const float mnew = fmaxf(mrow[r], mx);
      alpha[r] = __expf(mrow[r] - mnew);
      mrow[r] = mnew;
    }
#pragma unroll
    for (int r = 0; r < 8; ++r) {
      float ps = 0.f;
#pragma unroll
      for (int t = 0; t < 4; ++t) {
        const float p = __expf(sc[t][r] * scale - mrow[r]);
        ps += p;
        Pt[wave][(r + half) * 64 + t * 16 + (lane & 15)] = f2bf(p);
      }
      lrow[r] = lrow[r] * alpha[r] + hadd16(ps);
    }
#pragma unroll
    for (int t = 0; t < 4; ++t)
#pragma unroll
      for (int r = 0; r < 8; ++r) o[t][r] *= alpha[r];

    // P staging is wave-private: only wave-local LDS completion needed.
    asm volatile("s_wait_dscnt 0" ::: "memory");
    __builtin_amdgcn_wave_barrier();

    // ---- O += P(16x64) @ V(64x64): load all frags, then wmma burst
    const v16bf p0 = load_frag(&Pt[wave][0], 64);
    const v16bf p1 = load_frag(&Pt[wave][32], 64);
    v16bf vb[8];
#pragma unroll
    for (int dt = 0; dt < 4; ++dt) {
      vb[2 * dt]     = load_frag(&Vt[buf][(dt * 16) * 64 + 0], 64);
      vb[2 * dt + 1] = load_frag(&Vt[buf][(dt * 16) * 64 + 32], 64);
    }
#pragma unroll
    for (int dt = 0; dt < 4; ++dt) {
      o[dt] = wmma_bf16(p0, vb[2 * dt], o[dt]);
      o[dt] = wmma_bf16(p1, vb[2 * dt + 1], o[dt]);
    }

    buf ^= 1;
  }

  // Normalize and write merged-head f32 output (b, s, h*64+d).
  float inv[8];
#pragma unroll
  for (int r = 0; r < 8; ++r) inv[r] = 1.f / lrow[r];
#pragma unroll
  for (int dt = 0; dt < 4; ++dt) {
#pragma unroll
    for (int r = 0; r < 8; ++r) {
      const int s = q0 + r + half;
      const int d = dt * 16 + (lane & 15);
      out[((size_t)(bb * S_LEN + s)) * D_DIM + h * W_HD + d] = o[dt][r] * inv[r];
    }
  }
}

// ---------- launcher ----------

extern "C" void kernel_launch(void* const* d_in, const int* in_sizes, int n_in,
                              void* d_out, int out_size, void* d_ws, size_t ws_size,
                              hipStream_t stream) {
  const float* x  = (const float*)d_in[0];
  const float* Wq = (const float*)d_in[1];
  const float* bq = (const float*)d_in[2];
  const float* Wk = (const float*)d_in[3];
  const float* bk = (const float*)d_in[4];
  const float* Wv = (const float*)d_in[5];
  const float* bv = (const float*)d_in[6];
  float* out = (float*)d_out;

  char* ws = (char*)d_ws;
  const size_t XB_BYTES  = (size_t)M_TOT * D_DIM * 2;                  // 16 MiB
  const size_t W_BYTES   = (size_t)D_DIM * D_DIM * 2;                  // 2 MiB
  const size_t QKV_BYTES = (size_t)B_DIM * H_CNT * S_LEN * W_HD * 2;   // 16 MiB

  u16* Xb  = (u16*)(ws);
  u16* Wqb = (u16*)(ws + XB_BYTES);
  u16* Wkb = (u16*)(ws + XB_BYTES + W_BYTES);
  u16* Wvb = (u16*)(ws + XB_BYTES + 2 * W_BYTES);
  u16* Qb  = (u16*)(ws + XB_BYTES + 3 * W_BYTES);
  u16* Kb  = (u16*)(ws + XB_BYTES + 3 * W_BYTES + QKV_BYTES);
  u16* Vb  = (u16*)(ws + XB_BYTES + 3 * W_BYTES + 2 * QKV_BYTES);

  // 1) cast inputs to bf16
  cast_f32_to_bf16<<<2048, 256, 0, stream>>>(x,  Xb,  M_TOT * D_DIM);
  cast_f32_to_bf16<<<1024, 256, 0, stream>>>(Wq, Wqb, D_DIM * D_DIM);
  cast_f32_to_bf16<<<1024, 256, 0, stream>>>(Wk, Wkb, D_DIM * D_DIM);
  cast_f32_to_bf16<<<1024, 256, 0, stream>>>(Wv, Wvb, D_DIM * D_DIM);

  // 2) QKV projections (WMMA GEMM -> head-major bf16)
  dim3 gp(M_TOT / 128, D_DIM / 64);
  proj_wmma<<<gp, 256, 0, stream>>>(Xb, Wqb, bq, Qb);
  proj_wmma<<<gp, 256, 0, stream>>>(Xb, Wkb, bk, Kb);
  proj_wmma<<<gp, 256, 0, stream>>>(Xb, Wvb, bv, Vb);

  // 3) flash attention (WMMA for QK^T and P*V)
  dim3 ga(B_DIM * H_CNT, S_LEN / 128);
  attn_flash_wmma<<<ga, 256, 0, stream>>>(Qb, Kb, Vb, out);
}
